// CoordinatesFusion_43937515438604
// MI455X (gfx1250) — matrix operand order repaired and checked by reading, not compiled, not run on hardware
//
#include <hip/hip_runtime.h>
#include <cstddef>
#include <cstdint>

// ---------------------------------------------------------------------------
// CDNA5 / gfx1250 fused CoordinatesFusion pipeline, wave32 + WMMA bf16
// + async global->LDS staging (ASYNCcnt path).
//   B=8, T=2048, D=512, 3*D=1536.  Peak workspace use: 128 MiB.
// ---------------------------------------------------------------------------

typedef __bf16 bf16_t;
typedef __attribute__((ext_vector_type(16))) __bf16 v16bf;
typedef __attribute__((ext_vector_type(8)))  float  v8f;

#define BB 8
#define TT 2048
#define DD 512

__device__ inline v8f wmma_bf16(v16bf a, v16bf b, v8f c) {
  return __builtin_amdgcn_wmma_f32_16x16x32_bf16(
      /*neg_a=*/false, a, /*neg_b=*/false, b,
      /*c_mod=*/(short)0, c, /*reuse_a=*/false, /*reuse_b=*/false);
}

// Async global->LDS direct copy, 16B per enabled lane (ASYNCcnt tracked).
// LDS byte address = low 32 bits of the generic pointer (ISA 10.2 aperture).
__device__ inline void async_ld_b128(unsigned lds_addr, const void* g) {
  asm volatile("global_load_async_to_lds_b128 %0, %1, off"
               :: "v"(lds_addr), "v"((unsigned long long)(uintptr_t)g)
               : "memory");
}
__device__ inline void wait_async0() {
  asm volatile("s_wait_asynccnt 0x0" ::: "memory");
}

// Load a 16-lane-wide bf16 fragment: two 16B LDS reads from one 32-elem row.
__device__ inline v16bf frag_ld(const bf16_t* rowp, int o0, int o1) {
  union { v16bf v; uint4 u[2]; } f;
  f.u[0] = *reinterpret_cast<const uint4*>(rowp + o0);
  f.u[1] = *reinterpret_cast<const uint4*>(rowp + o1);
  return f.v;
}

__device__ inline float gelu_erf(float x) {
  return 0.5f * x * (1.0f + erff(x * 0.70710678118654752f));
}

// ---------------------------------------------------------------------------
// Generic tiled GEMM:  C[z] = epilogue(A[z] @ B[z] + bias)
//   A row-major [M x K] (lda), B row-major [K x N] (ldb), C row-major (ldc).
//   Block tile 128x64, K-step 32, 8 waves -> 32x32 per wave (2x2 WMMA tiles).
//   bf16 A tiles are staged with global_load_async_to_lds_b128 (layout is
//   identical in LDS and memory); fp32 A and all B tiles go through the
//   converting/transposing manual path.
// Epilogues: 0 = plain -> bf16
//            1 = bias + gelu -> bf16
//            2 = bias -> f32
//            3 = gelu(bias) + residual -> f32
// ---------------------------------------------------------------------------
template <typename TA, typename TB, int EPI>
__global__ __launch_bounds__(256)
void gemm_wmma(const TA* __restrict__ A, const TB* __restrict__ B,
               const float* __restrict__ bias, const float* __restrict__ res,
               void* __restrict__ Cv, int M, int N, int K,
               int lda, int ldb, int ldc,
               long long sA, long long sB, long long sC) {
  __shared__ bf16_t As[128 * 32];   // [row][k]  row stride 32
  __shared__ bf16_t Bs[64 * 32];    // [n][k]    (B staged transposed)

  const int tid  = threadIdx.x;
  const int lane = tid & 31;
  const int wave = tid >> 5;
  const int half = lane >> 4;       // 0: lanes 0-15, 1: lanes 16-31
  const int lrow = lane & 15;
  const int wm   = wave >> 1;       // 0..3 -> m offset wm*32
  const int wn   = wave & 1;        // 0..1 -> n offset wn*32

  const long long z = blockIdx.z;
  A += z * sA;
  B += z * sB;
  const size_t cbase = (size_t)(z * sC);
  const int row0 = blockIdx.y * 128;
  const int n0   = blockIdx.x * 64;

  v8f c[2][2];
#pragma unroll
  for (int i = 0; i < 2; ++i)
#pragma unroll
    for (int j = 0; j < 2; ++j)
#pragma unroll
      for (int v = 0; v < 8; ++v) c[i][j][v] = 0.0f;

  const unsigned as_base = (unsigned)(uintptr_t)&As[0];

  for (int k0 = 0; k0 < K; k0 += 32) {
    __syncthreads();
    // ---- Stage A tile 128x32 into As[r][k] ----
    if constexpr (sizeof(TA) == 2) {
      // bf16 source: direct async copy, 16B chunks. e = r*4 + c16.
#pragma unroll
      for (int i = 0; i < 2; ++i) {
        int e = tid + i * 256;
        int r = e >> 2, c16 = e & 3;
        async_ld_b128(as_base + (unsigned)e * 16,
                      A + (size_t)(row0 + r) * lda + k0 + c16 * 8);
      }
    } else {
      // fp32 source: load + convert. e = r*32 + kk, coalesced in kk.
#pragma unroll
      for (int i = 0; i < 16; ++i) {
        int e = tid + i * 256;
        int r = e >> 5, kk = e & 31;
        As[e] = (bf16_t)(float)A[(size_t)(row0 + r) * lda + k0 + kk];
      }
    }
    // ---- Stage B tile 64x32 transposed into Bs[n][k] ----
#pragma unroll
    for (int i = 0; i < 8; ++i) {
      int e = tid + i * 256;
      int k = e >> 6, n = e & 63;
      Bs[n * 32 + k] = (bf16_t)(float)B[(size_t)(k0 + k) * ldb + n0 + n];
    }
    // Prefetch next B k-slab into GL2 (global_prefetch_b8).
    if (k0 + 32 < K)
      __builtin_prefetch(&B[(size_t)(k0 + 32 + (tid >> 6)) * ldb + n0 + (tid & 63)], 0, 0);
    if constexpr (sizeof(TA) == 2) wait_async0();
    __syncthreads();

    v16bf a[2], b[2];
#pragma unroll
    for (int i = 0; i < 2; ++i)
      a[i] = frag_ld(&As[(wm * 32 + i * 16 + lrow) * 32], half * 16, half * 16 + 8);
#pragma unroll
    for (int j = 0; j < 2; ++j)
      b[j] = frag_ld(&Bs[(wn * 32 + j * 16 + lrow) * 32], half * 16, half * 16 + 8);
#pragma unroll
    for (int i = 0; i < 2; ++i)
#pragma unroll
      for (int j = 0; j < 2; ++j)
        c[i][j] = wmma_bf16(a[i], b[j], c[i][j]);
  }

  // Epilogue. C/D layout: lane -> N=lrow, VGPR v -> M = v + half*8.
  float*  Cf = (float*)Cv;
  bf16_t* Ch = (bf16_t*)Cv;
#pragma unroll
  for (int i = 0; i < 2; ++i)
#pragma unroll
    for (int j = 0; j < 2; ++j)
#pragma unroll
      for (int v = 0; v < 8; ++v) {
        int grow = row0 + wm * 32 + i * 16 + half * 8 + v;
        int gcol = n0 + wn * 32 + j * 16 + lrow;
        float val = c[i][j][v];
        if (EPI >= 1) val += bias[gcol];
        if (EPI == 1) val = gelu_erf(val);
        if (EPI == 3) val = gelu_erf(val) + res[(size_t)grow * ldc + gcol];
        size_t off = cbase + (size_t)grow * ldc + gcol;
        if (EPI <= 1) Ch[off] = (bf16_t)val;
        else          Cf[off] = val;
      }
}

// ---------------------------------------------------------------------------
// Fused attention scores + softmax:
//   P[b, q0..q0+15, :] = softmax_row( R[b,q,:] @ L[b,:, :]^T )
// One WG per (16-query block, batch). 8 waves; wave w owns key tiles
// ntile = p*32 + t*8 + w (p=phase 0..3, t=0..3) -> 16 accumulator tiles/wave.
// Keys staged 512 at a time via async global->LDS copies (34 KB LDS total).
// ---------------------------------------------------------------------------
__global__ __launch_bounds__(256)
void attn_scores_softmax(const bf16_t* __restrict__ R,   // right_out bf16 [B,T,D]
                         const bf16_t* __restrict__ L,   // left_out  bf16 [B,T,D]
                         bf16_t* __restrict__ P) {       // probs     bf16 [B,T,T]
  __shared__ bf16_t Aq[16 * 32];     // query slab [q][k]
  __shared__ bf16_t Bs[512 * 32];    // key slab   [key][k]
  __shared__ float  red1[8 * 16];
  __shared__ float  red2[8 * 16];

  const int tid   = threadIdx.x;
  const int lane  = tid & 31;
  const int w     = tid >> 5;
  const int half  = lane >> 4;
  const int lrow  = lane & 15;
  const int batch = blockIdx.y;
  const int qbase = blockIdx.x * 16;

  const size_t boff = (size_t)batch * TT * DD;
  const size_t qoff = boff + (size_t)qbase * DD;
  const unsigned aq_base = (unsigned)(uintptr_t)&Aq[0];
  const unsigned bs_base = (unsigned)(uintptr_t)&Bs[0];

  v8f c[16];
#pragma unroll
  for (int t = 0; t < 16; ++t)
#pragma unroll
    for (int v = 0; v < 8; ++v) c[t][v] = 0.0f;

  for (int k0 = 0; k0 < DD; k0 += 32) {
    __syncthreads();
    // Stage 16x32 query slab: 64 async 16B chunks (waves 0-1). e = r*4 + c16.
    if (tid < 64) {
      int r = tid >> 2, c16 = tid & 3;
      async_ld_b128(aq_base + (unsigned)tid * 16,
                    R + qoff + (size_t)r * DD + k0 + c16 * 8);
    }
#pragma unroll
    for (int p = 0; p < 4; ++p) {
      __syncthreads();
      // Stage 512 keys x 32 feats: 2048 async 16B chunks (8 per thread).
      const size_t kb = boff + (size_t)(p * 512) * DD + k0;
#pragma unroll
      for (int i = 0; i < 8; ++i) {
        int e = tid + i * 256;
        int key = e >> 2, c16 = e & 3;
        async_ld_b128(bs_base + (unsigned)e * 16,
                      L + kb + (size_t)key * DD + c16 * 8);
      }
      wait_async0();
      __syncthreads();
      v16bf a = frag_ld(&Aq[lrow * 32], half * 16, half * 16 + 8);
#pragma unroll
      for (int t = 0; t < 4; ++t) {
        int nrow = (t * 8 + w) * 16 + lrow;   // 0..511
        v16bf bfr = frag_ld(&Bs[nrow * 32], half * 16, half * 16 + 8);
        c[p * 4 + t] = wmma_bf16(a, bfr, c[p * 4 + t]);
      }
    }
  }

  // ---- softmax over each query row (2048 cols spread over waves) ----
  // Lane owns rows M = half*8 + v (v=0..7) at column lrow of each tile.
  float lm[8];
#pragma unroll
  for (int v = 0; v < 8; ++v) lm[v] = -3.0e38f;
#pragma unroll
  for (int t = 0; t < 16; ++t)
#pragma unroll
    for (int v = 0; v < 8; ++v) lm[v] = fmaxf(lm[v], c[t][v]);
  for (int m = 1; m < 16; m <<= 1)
#pragma unroll
    for (int v = 0; v < 8; ++v) lm[v] = fmaxf(lm[v], __shfl_xor(lm[v], m, 32));
  if (lrow == 0)
#pragma unroll
    for (int v = 0; v < 8; ++v) red1[w * 16 + half * 8 + v] = lm[v];
  __syncthreads();

  float fm[8];
#pragma unroll
  for (int v = 0; v < 8; ++v) {
    float m = red1[half * 8 + v];
#pragma unroll
    for (int w2 = 1; w2 < 8; ++w2) m = fmaxf(m, red1[w2 * 16 + half * 8 + v]);
    fm[v] = m;
  }

  float ls[8];
#pragma unroll
  for (int v = 0; v < 8; ++v) ls[v] = 0.0f;
#pragma unroll
  for (int t = 0; t < 16; ++t)
#pragma unroll
    for (int v = 0; v < 8; ++v) {
      c[t][v] = __expf(c[t][v] - fm[v]);
      ls[v] += c[t][v];
    }
  for (int m = 1; m < 16; m <<= 1)
#pragma unroll
    for (int v = 0; v < 8; ++v) ls[v] += __shfl_xor(ls[v], m, 32);
  if (lrow == 0)
#pragma unroll
    for (int v = 0; v < 8; ++v) red2[w * 16 + half * 8 + v] = ls[v];
  __syncthreads();

  float rinv[8];
#pragma unroll
  for (int v = 0; v < 8; ++v) {
    float s = 0.0f;
#pragma unroll
    for (int w2 = 0; w2 < 8; ++w2) s += red2[w2 * 16 + half * 8 + v];
    rinv[v] = 1.0f / s;
  }

  const size_t pbase = (size_t)batch * TT * TT;
#pragma unroll
  for (int p = 0; p < 4; ++p)
#pragma unroll
    for (int t = 0; t < 4; ++t)
#pragma unroll
      for (int v = 0; v < 8; ++v) {
        int row = qbase + half * 8 + v;
        int col = (p * 32 + t * 8 + w) * 16 + lrow;
        P[pbase + (size_t)row * TT + col] = (bf16_t)(c[p * 4 + t][v] * rinv[v]);
      }
}

// ---------------------------------------------------------------------------
// LayerNorm over rows of 512: one wave per row, 8 rows per 256-thread block.
// Emits bf16 image (next GEMM input); optionally fp32 image (residual path).
// ---------------------------------------------------------------------------
template <bool WRITE_F32>
__global__ __launch_bounds__(256)
void layernorm_rows(const float* __restrict__ X, const float* __restrict__ g,
                    const float* __restrict__ b, bf16_t* __restrict__ Yh,
                    float* __restrict__ Yf) {
  const int lane = threadIdx.x & 31;
  const int wave = threadIdx.x >> 5;
  const int row  = blockIdx.x * 8 + wave;

  const float4* xp = (const float4*)(X + (size_t)row * DD);
  float4 x[4];
  float s = 0.0f, ss = 0.0f;
#pragma unroll
  for (int i = 0; i < 4; ++i) {
    x[i] = xp[lane + i * 32];
    s  += x[i].x + x[i].y + x[i].z + x[i].w;
    ss += x[i].x * x[i].x + x[i].y * x[i].y + x[i].z * x[i].z + x[i].w * x[i].w;
  }
  for (int m = 1; m < 32; m <<= 1) {
    s  += __shfl_xor(s, m, 32);
    ss += __shfl_xor(ss, m, 32);
  }
  const float mean = s * (1.0f / DD);
  const float var  = ss * (1.0f / DD) - mean * mean;
  const float rstd = rsqrtf(var + 1e-5f);

#pragma unroll
  for (int i = 0; i < 4; ++i) {
    int c4 = lane + i * 32;
    float4 gv = ((const float4*)g)[c4];
    float4 bv = ((const float4*)b)[c4];
    float4 y;
    y.x = (x[i].x - mean) * rstd * gv.x + bv.x;
    y.y = (x[i].y - mean) * rstd * gv.y + bv.y;
    y.z = (x[i].z - mean) * rstd * gv.z + bv.z;
    y.w = (x[i].w - mean) * rstd * gv.w + bv.w;
    size_t o = (size_t)row * DD + (size_t)c4 * 4;
    Yh[o + 0] = (bf16_t)y.x;
    Yh[o + 1] = (bf16_t)y.y;
    Yh[o + 2] = (bf16_t)y.z;
    Yh[o + 3] = (bf16_t)y.w;
    if (WRITE_F32) ((float4*)(Yf + (size_t)row * DD))[c4] = y;
  }
}

// ---------------------------------------------------------------------------
// Host-side pipeline. Workspace aliasing (peak 128 MiB):
//   [0,16)    lo bf16            -> later h1ln bf16
//   [16,32)   ro bf16            -> later part of h2 bf16
//   [32,48)   bo bf16            -> later fuse_ln bf16 / part of h2
//   [48,112)  P  bf16            -> later fuse_ln f32 + h1 f32
//   [112,128) fuse0 bf16
//   [0,32)    fuse1 f32 (after lo/ro dead)
// ---------------------------------------------------------------------------
extern "C" void kernel_launch(void* const* d_in, const int* in_sizes, int n_in,
                              void* d_out, int out_size, void* d_ws, size_t ws_size,
                              hipStream_t stream) {
  (void)in_sizes; (void)n_in; (void)out_size; (void)ws_size;
  const float* left  = (const float*)d_in[0];
  const float* right = (const float*)d_in[1];
  const float* body  = (const float*)d_in[2];
  const float* Wl = (const float*)d_in[3];  const float* bl = (const float*)d_in[4];
  const float* Wr = (const float*)d_in[5];  const float* br = (const float*)d_in[6];
  const float* Wb = (const float*)d_in[7];  const float* bb = (const float*)d_in[8];
  const float* Wo = (const float*)d_in[9];  const float* bo_b = (const float*)d_in[10];
  const float* ln_g = (const float*)d_in[11]; const float* ln_b = (const float*)d_in[12];
  const float* W1 = (const float*)d_in[13]; const float* b1 = (const float*)d_in[14];
  const float* ilg = (const float*)d_in[15]; const float* ilb = (const float*)d_in[16];
  const float* W2 = (const float*)d_in[17]; const float* b2 = (const float*)d_in[18];
  const float* W3 = (const float*)d_in[19]; const float* b3 = (const float*)d_in[20];

  constexpr size_t MiB = 1024u * 1024u;
  char* w = (char*)d_ws;
  bf16_t* lo    = (bf16_t*)(w + 0 * MiB);
  bf16_t* ro    = (bf16_t*)(w + 16 * MiB);
  bf16_t* bo    = (bf16_t*)(w + 32 * MiB);
  bf16_t* P     = (bf16_t*)(w + 48 * MiB);
  bf16_t* fuse0 = (bf16_t*)(w + 112 * MiB);
  float*  fuse1 = (float*) (w + 0 * MiB);
  bf16_t* lnh   = (bf16_t*)(w + 32 * MiB);
  float*  lnf   = (float*) (w + 48 * MiB);
  float*  h1    = (float*) (w + 80 * MiB);
  bf16_t* h1h   = (bf16_t*)(w + 0 * MiB);
  bf16_t* h2    = (bf16_t*)(w + 16 * MiB);

  const int M = BB * TT;   // 16384
  dim3 blk(256);

  // 1-3: input projections + GELU -> bf16
  gemm_wmma<float, float, 1><<<dim3(DD / 64, M / 128, 1), blk, 0, stream>>>(
      left, Wl, bl, nullptr, lo, M, DD, DD, DD, DD, DD, 0, 0, 0);
  gemm_wmma<float, float, 1><<<dim3(DD / 64, M / 128, 1), blk, 0, stream>>>(
      right, Wr, br, nullptr, ro, M, DD, DD, DD, DD, DD, 0, 0, 0);
  gemm_wmma<float, float, 1><<<dim3(DD / 64, M / 128, 1), blk, 0, stream>>>(
      body, Wb, bb, nullptr, bo, M, DD, DD, DD, DD, DD, 0, 0, 0);

  // 4: fused scores + softmax -> P bf16
  attn_scores_softmax<<<dim3(TT / 16, BB), blk, 0, stream>>>(ro, lo, P);

  // 5: fuse0 = P @ body_out   (batched over grid.z)
  gemm_wmma<bf16_t, bf16_t, 0><<<dim3(DD / 64, TT / 128, BB), blk, 0, stream>>>(
      P, bo, nullptr, nullptr, fuse0, TT, DD, TT, TT, DD, DD,
      (long long)TT * TT, (long long)TT * DD, (long long)TT * DD);

  // 6: fuse1 = fuse0 @ Wo + bo -> f32
  gemm_wmma<bf16_t, float, 2><<<dim3(DD / 64, M / 128, 1), blk, 0, stream>>>(
      fuse0, Wo, bo_b, nullptr, fuse1, M, DD, DD, DD, DD, DD, 0, 0, 0);

  // 7: LayerNorm -> bf16 + f32 (residual source)
  layernorm_rows<true><<<dim3(M / 8), blk, 0, stream>>>(fuse1, ln_g, ln_b, lnh, lnf);

  // 8: h1 = gelu(lnh @ W1 + b1) + lnf -> f32
  gemm_wmma<bf16_t, float, 3><<<dim3(DD / 64, M / 128, 1), blk, 0, stream>>>(
      lnh, W1, b1, lnf, h1, M, DD, DD, DD, DD, DD, 0, 0, 0);

  // 9: LayerNorm -> bf16
  layernorm_rows<false><<<dim3(M / 8), blk, 0, stream>>>(h1, ilg, ilb, h1h, nullptr);

  // 10: h2 = gelu(h1h @ W2 + b2) -> bf16   [M x 1536]
  gemm_wmma<bf16_t, float, 1><<<dim3(1536 / 64, M / 128, 1), blk, 0, stream>>>(
      h1h, W2, b2, nullptr, h2, M, 1536, DD, DD, 1536, 1536, 0, 0, 0);

  // 11: out = h2 @ W3 + b3 -> f32 (final output)
  gemm_wmma<bf16_t, float, 2><<<dim3(DD / 64, M / 128, 1), blk, 0, stream>>>(
      h2, W3, b3, nullptr, (float*)d_out, M, DD, 1536, 1536, DD, DD, 0, 0, 0);
}
